// Agent_71777493451340
// MI455X (gfx1250) — matrix operand order repaired and checked by reading, not compiled, not run on hardware
//
#include <hip/hip_runtime.h>
#include <hip/hip_bf16.h>

// ---------------- problem constants (match reference) ----------------
constexpr int kT   = 64;
constexpr int kB   = 2048;
constexpr int kOBS = 48;
constexpr int kH   = 256;      // hidden
constexpr int kG4  = 4 * kH;   // 1024 gate columns
constexpr int kM1  = 512;
constexpr int kM2  = 256;
constexpr int kA   = 12;
constexpr int kAp  = 16;       // padded head width
constexpr int kKX  = 64;       // OBS padded to 64 for K=32 WMMA steps
constexpr float kLOG2PI = 1.8378770664093453f;

// ---------------- WMMA fragment types ----------------
typedef __attribute__((ext_vector_type(16))) __bf16 v16bf;
typedef __attribute__((ext_vector_type(8)))  float  v8f;

union FragAB {
    v16bf v;
    uint4 q[2];
    unsigned short s[16];
};

__device__ __forceinline__ unsigned short f2bf(float f) {
    unsigned int u = __float_as_uint(f);
    unsigned int r = u + 0x7fffu + ((u >> 16) & 1u);   // round-to-nearest-even
    return (unsigned short)(r >> 16);
}
__device__ __forceinline__ float bf2f(unsigned short s) {
    return __uint_as_float(((unsigned int)s) << 16);
}
__device__ __forceinline__ float sigmoid_f(float x) { return 1.0f / (1.0f + __expf(-x)); }
__device__ __forceinline__ float tanh_f(float x) {
    float e = __expf(2.0f * x);
    return (e - 1.0f) / (e + 1.0f);
}
__device__ __forceinline__ float elu_f(float x) { return x > 0.0f ? x : (__expf(x) - 1.0f); }

__device__ __forceinline__ v8f wmma_bf16(const FragAB& a, const FragAB& b, v8f c) {
    return __builtin_amdgcn_wmma_f32_16x16x32_bf16(false, a.v, false, b.v,
                                                   (short)0, c, false, false);
}

// ---------------- weight prep: fp32 -> bf16, optional transpose + pad ----------------
__global__ void k_prep(const float* __restrict__ src, unsigned short* __restrict__ dst,
                       int Ndst, int Kdst, int Nv, int Kv, int srcld, int transpose) {
    int i = blockIdx.x * blockDim.x + threadIdx.x;
    int total = Ndst * Kdst;
    if (i >= total) return;
    int n = i / Kdst, k = i - n * Kdst;
    float v = 0.0f;
    if (transpose) {
        if (n < Nv && k < Kv) v = src[k * srcld + n];
    } else {
        if (k < Kv) v = src[n * Kv + k];
    }
    dst[i] = f2bf(v);
}

__global__ void k_bias_sum(const float* __restrict__ a, const float* __restrict__ b,
                           float* __restrict__ o, int n) {
    int i = blockIdx.x * blockDim.x + threadIdx.x;
    if (i < n) o[i] = a[i] + b[i];
}

// ---------------- fused LSTM scan (x-projection + recurrence + cell) ----------------
// grid = B/32 workgroups, 1024 threads (32 waves). Each WG owns 32 batch rows for all T
// (halves L2 weight traffic vs 16-row tiles: one W_hh stream feeds 32 rows).
// Wave w: row-half mt = w>>4 (rows mt*16..mt*16+15), gate-column tile (w&15)*16 of each
// of the 4 gates -> acc[4] v8f = 32 VGPRs per lane.
__global__ __launch_bounds__(1024) void k_scan(
    const float* __restrict__ x,            // [T,B,48]
    const int*   __restrict__ done,         // [T,B]
    const float* __restrict__ h0,           // [B,256]
    const float* __restrict__ c0,           // [B,256]
    const unsigned short* __restrict__ Wih, // [1024,64]  bf16, [g][k]
    const unsigned short* __restrict__ Whh, // [1024,256] bf16, [g][k]
    const float* __restrict__ bias,         // [1024] = b_ih + b_hh
    unsigned short* __restrict__ hs)        // [T*B,256] bf16 out
{
    constexpr int BT = 32;
    constexpr int NT = 1024;        // threads
    constexpr int HB_ST = kH + 8;   // 264: padded stride (banks)
    constexpr int XB_ST = kKX + 8;  // 72

    __shared__ unsigned short h_bf[BT * HB_ST];   // 16.5 KB
    __shared__ unsigned short x_bf[BT * XB_ST];   //  4.5 KB
    __shared__ float bias_s[kG4];                 //  4   KB

    const int tid   = threadIdx.x;
    const int lane  = tid & 31;
    const int wave  = tid >> 5;        // 0..31
    const int half  = lane >> 4;       // 0/1
    const int l15   = lane & 15;
    const int mt    = wave >> 4;       // row-half 0/1
    const int bbase = blockIdx.x * BT;
    const int jcol  = (wave & 15) * 16 + l15; // hidden column owned (per gate)
    const int arow  = mt * 16 + l15;          // A-fragment row in the 32-row tile

    for (int i = tid; i < kG4; i += NT) bias_s[i] = bias[i];
    for (int i = tid; i < BT * kH; i += NT) {
        int b = i >> 8, k = i & 255;
        h_bf[b * HB_ST + k] = f2bf(h0[(bbase + b) * kH + k]);
    }

    // c in registers, laid out exactly like the WMMA D fragment (m = mt*16+r+8*half, n = jcol)
    float creg[8];
    #pragma unroll
    for (int r = 0; r < 8; ++r) {
        int m = mt * 16 + r + 8 * half;
        creg[r] = c0[(bbase + m) * kH + jcol];
    }
    __syncthreads();

    for (int t = 0; t < kT; ++t) {
        // apply episode-reset mask to h (in place), stage x tile (bf16, K padded)
        for (int i = tid; i < BT * kH; i += NT) {
            int b = i >> 8, k = i & 255;
            float kp = 1.0f - (float)done[t * kB + bbase + b];
            unsigned short* p = &h_bf[b * HB_ST + k];
            *p = f2bf(bf2f(*p) * kp);
        }
        for (int i = tid; i < BT * kKX; i += NT) {
            int b = i >> 6, k = i & 63;
            float v = 0.0f;
            if (k < kOBS) {
                v = x[((size_t)t * kB + bbase + b) * kOBS + k];
                if (t + 1 < kT)   // stream-prefetch next timestep's observation tile
                    __builtin_prefetch(&x[((size_t)(t + 1) * kB + bbase + b) * kOBS + k], 0, 1);
            }
            x_bf[b * XB_ST + k] = f2bf(v);
        }
        __syncthreads();

        // gate accumulators initialized with bias (one 16-col tile per gate)
        v8f acc[4];
        #pragma unroll
        for (int g = 0; g < 4; ++g) {
            float bv = bias_s[g * kH + jcol];
            v8f a;
            #pragma unroll
            for (int r = 0; r < 8; ++r) a[r] = bv;
            acc[g] = a;
        }

        // recurrent GEMM: h[32,256] x Whh^T -> gates, K = 256
        #pragma unroll 1
        for (int k0 = 0; k0 < kH; k0 += 32) {
            FragAB af;
            const unsigned short* ap = &h_bf[arow * HB_ST + k0 + half * 8];
            af.q[0] = *(const uint4*)ap;
            af.q[1] = *(const uint4*)(ap + 16);
            #pragma unroll
            for (int g = 0; g < 4; ++g) {
                FragAB bfr;
                const unsigned short* bp = &Whh[(g * kH + jcol) * kH + k0 + half * 8];
                bfr.q[0] = *(const uint4*)bp;
                bfr.q[1] = *(const uint4*)(bp + 16);
                acc[g] = wmma_bf16(af, bfr, acc[g]);
            }
        }
        // input-side GEMM: x[32,64] x Wih^T, K = 64 (padded)
        #pragma unroll 1
        for (int k0 = 0; k0 < kKX; k0 += 32) {
            FragAB af;
            const unsigned short* ap = &x_bf[arow * XB_ST + k0 + half * 8];
            af.q[0] = *(const uint4*)ap;
            af.q[1] = *(const uint4*)(ap + 16);
            #pragma unroll
            for (int g = 0; g < 4; ++g) {
                FragAB bfr;
                const unsigned short* bp = &Wih[(g * kH + jcol) * kKX + k0 + half * 8];
                bfr.q[0] = *(const uint4*)bp;
                bfr.q[1] = *(const uint4*)(bp + 16);
                acc[g] = wmma_bf16(af, bfr, acc[g]);
            }
        }
        __syncthreads();   // all reads of h_bf complete before overwrite

        // LSTM cell update (register-local; D-fragment layout)
        #pragma unroll
        for (int r = 0; r < 8; ++r) {
            int m = mt * 16 + r + 8 * half;
            float kp = 1.0f - (float)done[t * kB + bbase + m];
            float ig = sigmoid_f(acc[0][r]);
            float fg = sigmoid_f(acc[1][r]);
            float gg = tanh_f(acc[2][r]);
            float og = sigmoid_f(acc[3][r]);
            float c  = fg * (creg[r] * kp) + ig * gg;
            creg[r] = c;
            float h = og * tanh_f(c);
            unsigned short hb = f2bf(h);
            h_bf[m * HB_ST + jcol] = hb;
            hs[((size_t)t * kB + bbase + m) * kH + jcol] = hb;
        }
        __syncthreads();
    }
}

// ---------------- LayerNorm, wave-per-row, in place on bf16 hs ----------------
__global__ __launch_bounds__(256) void k_ln(unsigned short* __restrict__ hs,
                                            const float* __restrict__ g,
                                            const float* __restrict__ b) {
    const size_t row = (size_t)blockIdx.x * 8 + (threadIdx.x >> 5);
    const int lane = threadIdx.x & 31;
    unsigned short* rp = hs + row * kH + lane * 8;
    uint4 q = *(const uint4*)rp;
    unsigned short* s = (unsigned short*)&q;
    float v[8], s1 = 0.f, s2 = 0.f;
    #pragma unroll
    for (int i = 0; i < 8; ++i) { v[i] = bf2f(s[i]); s1 += v[i]; s2 += v[i] * v[i]; }
    #pragma unroll
    for (int m = 16; m >= 1; m >>= 1) {
        s1 += __shfl_xor(s1, m, 32);
        s2 += __shfl_xor(s2, m, 32);
    }
    float mu = s1 * (1.0f / kH);
    float var = s2 * (1.0f / kH) - mu * mu;
    float rs = rsqrtf(var + 1e-5f);
    #pragma unroll
    for (int i = 0; i < 8; ++i) {
        float y = (v[i] - mu) * rs * g[lane * 8 + i] + b[lane * 8 + i];
        s[i] = f2bf(y);
    }
    *(uint4*)rp = q;
}

// ---------------- fused actor MLP + heads + logp/entropy ----------------
// grid = (T*B)/32 workgroups, 256 threads. Activations stay in LDS.
// K loops are outside N-tile loops so one A-fragment feeds 4 (layer1) / 2 (layer2) WMMAs.
__global__ __launch_bounds__(256) void k_mlp(
    const unsigned short* __restrict__ y,    // [R,256] bf16 (normalized)
    const unsigned short* __restrict__ W1t,  // [512,256]
    const float* __restrict__ b1,
    const unsigned short* __restrict__ W2t,  // [256,512]
    const float* __restrict__ b2,
    const unsigned short* __restrict__ Wmt,  // [16,256]
    const float* __restrict__ bm,
    const unsigned short* __restrict__ Wst,  // [16,256]
    const float* __restrict__ bs,
    float* __restrict__ out)                 // [R,14]
{
    constexpr int MT = 32;
    constexpr int A1_ST = kM1 + 8;   // 520
    constexpr int A2_ST = kM2 + 8;   // 264
    __shared__ unsigned short a1s[MT * A1_ST];
    __shared__ unsigned short a2s[MT * A2_ST];
    __shared__ float mean_s[MT * kAp];
    __shared__ float ls_s[MT * kAp];

    const int tid  = threadIdx.x;
    const int lane = tid & 31;
    const int wave = tid >> 5;
    const int half = lane >> 4;
    const int l15  = lane & 15;
    const size_t rowbase = (size_t)blockIdx.x * MT;

    // ---- layer 1: y[32,256] @ W1 -> ELU -> a1s[32,512]; wave owns 64 cols x both row-halves
    {
        const int n0 = wave * 64;
        #pragma unroll
        for (int mtl = 0; mtl < 2; ++mtl) {
            v8f acc[4];
            #pragma unroll
            for (int nt = 0; nt < 4; ++nt) {
                float bv = b1[n0 + nt * 16 + l15];
                #pragma unroll
                for (int r = 0; r < 8; ++r) acc[nt][r] = bv;
            }
            #pragma unroll 1
            for (int k0 = 0; k0 < kH; k0 += 32) {
                FragAB af;
                const unsigned short* ap =
                    y + (rowbase + mtl * 16 + l15) * kH + k0 + half * 8;
                af.q[0] = *(const uint4*)ap;
                af.q[1] = *(const uint4*)(ap + 16);
                #pragma unroll
                for (int nt = 0; nt < 4; ++nt) {
                    FragAB bfr;
                    const unsigned short* bp =
                        W1t + (n0 + nt * 16 + l15) * kH + k0 + half * 8;
                    bfr.q[0] = *(const uint4*)bp;
                    bfr.q[1] = *(const uint4*)(bp + 16);
                    acc[nt] = wmma_bf16(af, bfr, acc[nt]);
                }
            }
            #pragma unroll
            for (int nt = 0; nt < 4; ++nt)
                #pragma unroll
                for (int r = 0; r < 8; ++r)
                    a1s[(mtl * 16 + r + 8 * half) * A1_ST + n0 + nt * 16 + l15] =
                        f2bf(elu_f(acc[nt][r]));
        }
    }
    __syncthreads();

    // ---- layer 2: a1[32,512] @ W2 -> ELU -> a2s[32,256]; wave owns 32 cols x both halves
    {
        const int n0 = wave * 32;
        #pragma unroll
        for (int mtl = 0; mtl < 2; ++mtl) {
            v8f acc[2];
            #pragma unroll
            for (int nt = 0; nt < 2; ++nt) {
                float bv = b2[n0 + nt * 16 + l15];
                #pragma unroll
                for (int r = 0; r < 8; ++r) acc[nt][r] = bv;
            }
            #pragma unroll 1
            for (int k0 = 0; k0 < kM1; k0 += 32) {
                FragAB af;
                const unsigned short* ap =
                    &a1s[(mtl * 16 + l15) * A1_ST + k0 + half * 8];
                af.q[0] = *(const uint4*)ap;
                af.q[1] = *(const uint4*)(ap + 16);
                #pragma unroll
                for (int nt = 0; nt < 2; ++nt) {
                    FragAB bfr;
                    const unsigned short* bp =
                        W2t + (n0 + nt * 16 + l15) * kM1 + k0 + half * 8;
                    bfr.q[0] = *(const uint4*)bp;
                    bfr.q[1] = *(const uint4*)(bp + 16);
                    acc[nt] = wmma_bf16(af, bfr, acc[nt]);
                }
            }
            #pragma unroll
            for (int nt = 0; nt < 2; ++nt)
                #pragma unroll
                for (int r = 0; r < 8; ++r)
                    a2s[(mtl * 16 + r + 8 * half) * A2_ST + n0 + nt * 16 + l15] =
                        f2bf(elu_f(acc[nt][r]));
        }
    }
    __syncthreads();

    // ---- heads: mean (waves 0-1) and logstd (waves 2-3), N padded to 16
    if (wave < 4) {
        const int mtl = wave & 1;
        const bool is_mean = wave < 2;
        const unsigned short* Wt = is_mean ? Wmt : Wst;
        const float* bb = is_mean ? bm : bs;
        const int ncol = l15;
        v8f acc; float bv = (ncol < kA) ? bb[ncol] : 0.0f;
        #pragma unroll
        for (int r = 0; r < 8; ++r) acc[r] = bv;
        #pragma unroll 1
        for (int k0 = 0; k0 < kM2; k0 += 32) {
            FragAB af, bfr;
            const unsigned short* ap = &a2s[(mtl * 16 + l15) * A2_ST + k0 + half * 8];
            af.q[0] = *(const uint4*)ap;
            af.q[1] = *(const uint4*)(ap + 16);
            const unsigned short* bp = Wt + ncol * kM2 + k0 + half * 8;
            bfr.q[0] = *(const uint4*)bp;
            bfr.q[1] = *(const uint4*)(bp + 16);
            acc = wmma_bf16(af, bfr, acc);
        }
        float* dst = is_mean ? mean_s : ls_s;
        #pragma unroll
        for (int r = 0; r < 8; ++r)
            dst[(mtl * 16 + r + 8 * half) * kAp + ncol] = acc[r];
    }
    __syncthreads();

    // ---- finalize: clip logstd, logp, entropy, write [mean|logp|ent]
    if (tid < MT) {
        float lp = 0.f, en = 0.f;
        float* o = out + (rowbase + tid) * (kA + 2);
        #pragma unroll
        for (int a = 0; a < kA; ++a) {
            float ls = ls_s[tid * kAp + a];
            ls = fminf(fmaxf(ls, -5.0f), 2.0f);
            lp += -ls - 0.5f * kLOG2PI;
            en += 0.5f + 0.5f * kLOG2PI + ls;
            o[a] = mean_s[tid * kAp + a];
        }
        o[kA]     = lp;
        o[kA + 1] = en;
    }
}

// ---------------- host launch ----------------
extern "C" void kernel_launch(void* const* d_in, const int* in_sizes, int n_in,
                              void* d_out, int out_size, void* d_ws, size_t ws_size,
                              hipStream_t stream) {
    const float* x    = (const float*)d_in[0];
    const int*   done = (const int*)  d_in[1];
    const float* h0   = (const float*)d_in[2];
    const float* c0   = (const float*)d_in[3];
    const float* W_ih = (const float*)d_in[4];   // [1024,48]
    const float* W_hh = (const float*)d_in[5];   // [1024,256]
    const float* b_ih = (const float*)d_in[6];
    const float* b_hh = (const float*)d_in[7];
    const float* ln_g = (const float*)d_in[8];
    const float* ln_b = (const float*)d_in[9];
    const float* W1   = (const float*)d_in[10];  // [256,512]
    const float* b1   = (const float*)d_in[11];
    const float* W2   = (const float*)d_in[12];  // [512,256]
    const float* b2   = (const float*)d_in[13];
    const float* Wm   = (const float*)d_in[14];  // [256,12]
    const float* bm   = (const float*)d_in[15];
    const float* Ws   = (const float*)d_in[16];  // [256,12]
    const float* bs   = (const float*)d_in[17];
    float* out = (float*)d_out;

    // workspace layout (256B aligned chunks)
    char* ws = (char*)d_ws;
    size_t off = 0;
    auto take = [&](size_t bytes) { char* p = ws + off; off = (off + bytes + 255) & ~(size_t)255; return p; };
    unsigned short* Wih_bf = (unsigned short*)take((size_t)kG4 * kKX * 2);   // 128 KB
    unsigned short* Whh_bf = (unsigned short*)take((size_t)kG4 * kH * 2);    // 512 KB
    unsigned short* W1t_bf = (unsigned short*)take((size_t)kM1 * kH * 2);    // 256 KB
    unsigned short* W2t_bf = (unsigned short*)take((size_t)kM2 * kM1 * 2);   // 256 KB
    unsigned short* Wmt_bf = (unsigned short*)take((size_t)kAp * kM2 * 2);   // 8 KB
    unsigned short* Wst_bf = (unsigned short*)take((size_t)kAp * kM2 * 2);   // 8 KB
    float*          bias_g = (float*)take((size_t)kG4 * 4);                  // 4 KB
    unsigned short* hs_bf  = (unsigned short*)take((size_t)kT * kB * kH * 2); // 64 MB
    (void)ws_size; (void)in_sizes; (void)n_in; (void)out_size;

    auto g1 = [](int total) { return dim3((total + 255) / 256); };

    // weight prep (bf16, fragment-friendly [N][K] layouts)
    k_prep<<<g1(kG4 * kKX), 256, 0, stream>>>(W_ih, Wih_bf, kG4, kKX, kG4, kOBS, 0, 0);
    k_prep<<<g1(kG4 * kH ), 256, 0, stream>>>(W_hh, Whh_bf, kG4, kH,  kG4, kH,   0, 0);
    k_prep<<<g1(kM1 * kH ), 256, 0, stream>>>(W1,   W1t_bf, kM1, kH,  kM1, kH,   kM1, 1);
    k_prep<<<g1(kM2 * kM1), 256, 0, stream>>>(W2,   W2t_bf, kM2, kM1, kM2, kM1,  kM2, 1);
    k_prep<<<g1(kAp * kM2), 256, 0, stream>>>(Wm,   Wmt_bf, kAp, kM2, kA,  kM2,  kA,  1);
    k_prep<<<g1(kAp * kM2), 256, 0, stream>>>(Ws,   Wst_bf, kAp, kM2, kA,  kM2,  kA,  1);
    k_bias_sum<<<g1(kG4), 256, 0, stream>>>(b_ih, b_hh, bias_g, kG4);

    // fused LSTM scan (1024 threads = 32 waves per WG, 32-row batch tiles)
    k_scan<<<dim3(kB / 32), 1024, 0, stream>>>(x, done, h0, c0,
                                               Wih_bf, Whh_bf, bias_g, hs_bf);
    // LayerNorm in place
    k_ln<<<dim3(kT * kB / 8), 256, 0, stream>>>(hs_bf, ln_g, ln_b);
    // fused actor MLP + heads
    k_mlp<<<dim3(kT * kB / 32), 256, 0, stream>>>(hs_bf, W1t_bf, b1, W2t_bf, b2,
                                                  Wmt_bf, bm, Wst_bf, bs, out);
}
